// TrajPredictor_51616916963852
// MI455X (gfx1250) — matrix operand order repaired and checked by reading, not compiled, not run on hardware
//
#include <hip/hip_runtime.h>
#include <hip/hip_bf16.h>

// ---------------- constants from the reference ----------------
#define BB   128          // batch
#define TT   64           // encoder time steps
#define AA   32           // agents
#define FF   6            // features
#define HH   64           // encoder hidden
#define DD   12           // decoder hidden
#define BA   4096         // BB*AA independent sequences
#define G3H  192          // 3*HH
#define EH   128          // 2*HH enc_out feature dim

typedef __attribute__((ext_vector_type(16))) _Float16 v16h;
typedef __attribute__((ext_vector_type(8)))  float    v8f;

union V16 { v16h v; _Float16 h[16]; float4 f4[2]; };
union V8F { v8f  v; float f[8]; };

// branch-free fast gate nonlinearities (v_exp_f32 + v_rcp_f32)
__device__ __forceinline__ float fast_sigmoid(float x) {
  float e = __builtin_amdgcn_exp2f(-1.4426950408889634f * x);
  return __builtin_amdgcn_rcpf(1.f + e);
}
__device__ __forceinline__ float fast_tanh(float x) {
  float ax = fabsf(x);
  float u  = __builtin_amdgcn_exp2f(-2.8853900817779268f * ax);   // e^(-2|x|)
  float t  = (1.f - u) * __builtin_amdgcn_rcpf(1.f + u);
  return copysignf(t, x);
}

// A-matrix (16x32 f16) element->K map per ISA table:
// lanes 0-15:  elems 0..7 -> K=0..7,   elems 8..15 -> K=16..23
// lanes 16-31: elems 0..7 -> K=8..15,  elems 8..15 -> K=24..31
__device__ __forceinline__ int kmapA(int e, int hi) {
  int k = (e < 8) ? e : (e + 8);
  return k + (hi ? 8 : 0);
}
// B-matrix (32x16 f16): same K map as A (lanes 0-15: K=0..7,16..23; lanes 16-31: K=8..15,24..31)
__device__ __forceinline__ int kmapB(int e, int hi) { return ((e < 8) ? e : (e + 8)) + (hi ? 8 : 0); }

// ---------------- prep: f32 weights -> padded f16 ----------------
__global__ void prep_kernel(const float* __restrict__ Whh_f, const float* __restrict__ Whh_b,
                            const float* __restrict__ Wih_f, const float* __restrict__ Wih_b,
                            const float* __restrict__ W_attn,
                            _Float16* Whh16f, _Float16* Whh16b,
                            _Float16* Wih16f, _Float16* Wih16b,
                            _Float16* Wattn16) {
  int i = blockIdx.x * 256 + threadIdx.x;
  if (i < G3H * HH) { Whh16f[i] = (_Float16)Whh_f[i]; Whh16b[i] = (_Float16)Whh_b[i]; }
  if (i < G3H * 8) {                       // Wih padded K: 6 -> 8
    int j = i >> 3, k = i & 7;
    _Float16 zf = (_Float16)0.f;
    Wih16f[i] = (k < FF) ? (_Float16)Wih_f[j * FF + k] : zf;
    Wih16b[i] = (k < FF) ? (_Float16)Wih_b[j * FF + k] : zf;
  }
  if (i < 16 * EH) {                       // W_attn enc part, N padded 12 -> 16
    int d = i >> 7, k = i & 127;
    Wattn16[i] = (d < DD) ? (_Float16)W_attn[d * (2 * HH + DD) + DD + k] : (_Float16)0.f;
  }
}

// ---------------- bidirectional GRU encoder ----------------
// grid = (BA/16, 2); block = 128 (4 waves). Wave w owns cols [16w,16w+16) of each gate.
__global__ __launch_bounds__(128) void enc_kernel(
    const float* __restrict__ src,
    const _Float16* __restrict__ Whh16f, const _Float16* __restrict__ Whh16b,
    const _Float16* __restrict__ Wih16f, const _Float16* __restrict__ Wih16b,
    const float* __restrict__ bih_f, const float* __restrict__ bhh_f,
    const float* __restrict__ bih_b, const float* __restrict__ bhh_b,
    _Float16* __restrict__ enc16) {
  const int wv = threadIdx.x >> 5;
  const int ln = threadIdx.x & 31;
  const int hi = ln >> 4;
  const int nl = ln & 15;
  const int rowbase = blockIdx.x * 16;
  const int dir = blockIdx.y;

  const _Float16* Whh16 = dir ? Whh16b : Whh16f;
  const _Float16* Wih16 = dir ? Wih16b : Wih16f;
  const float*    bih   = dir ? bih_b  : bih_f;
  const float*    bhh   = dir ? bhh_b  : bhh_f;

  __shared__ _Float16 hbuf[16 * HH];
  for (int i = threadIdx.x; i < 16 * HH; i += 128) hbuf[i] = (_Float16)0.f;

  // pre-pack B fragments (kept in registers across all 64 steps)
  V16 bh[3][2];   // h-path weights: tiles {wv, wv+4, wv+8} x ksteps {0,1}
  V16 bx[3];      // x-path weights (K padded to 32, valid K<6)
  #pragma unroll
  for (int g = 0; g < 3; ++g) {
    const int tile = wv + 4 * g;
    const int j = tile * 16 + nl;          // output column (row of W)
    #pragma unroll
    for (int ks = 0; ks < 2; ++ks)
      #pragma unroll
      for (int e = 0; e < 16; ++e)
        bh[g][ks].h[e] = Whh16[j * HH + 32 * ks + kmapB(e, hi)];
    #pragma unroll
    for (int e = 0; e < 16; ++e) {
      int k = kmapB(e, hi);
      bx[g].h[e] = (k < 8) ? Wih16[j * 8 + k] : (_Float16)0.f;
    }
  }

  const int gcol = 16 * wv + nl;           // gate-local column this lane handles
  const float bias_r  = bih[gcol]            + bhh[gcol];
  const float bias_z  = bih[HH + gcol]       + bhh[HH + gcol];
  const float bias_ni = bih[2 * HH + gcol];
  const float bias_nh = bhh[2 * HH + gcol];

  // this lane's slice of the recurrent state (C/D layout: row m=v+8*hi, col=gcol)
  float hreg[8];
  #pragma unroll
  for (int v = 0; v < 8; ++v) hreg[v] = 0.f;

  // x source pointer for this lane's A-row (row index = nl; only lanes 0-15 carry x)
  const int xrow = rowbase + nl;
  const int xb = xrow >> 5, xa = xrow & 31;
  const float* xbase = src + (size_t)xb * TT * (AA * FF) + xa * FF;

  // software-pipelined x load (unconditional, contiguous, branch-free)
  float xv[6], xn[6];
  {
    const int t0 = dir ? (TT - 1) : 0;
    const float* xp = xbase + (size_t)t0 * (AA * FF);
    #pragma unroll
    for (int k = 0; k < FF; ++k) xv[k] = xp[k];
  }

  __syncthreads();

  #pragma unroll 1
  for (int step = 0; step < TT; ++step) {
    const int t = dir ? (TT - 1 - step) : step;

    // A fragment for x from registers; only lanes 0-15 (hi==0), elems 0..5 are nonzero
    V16 axf;
    #pragma unroll
    for (int e = 0; e < 16; ++e) {
      if (e < FF) axf.h[e] = hi ? (_Float16)0.f : (_Float16)xv[e];
      else        axf.h[e] = (_Float16)0.f;
    }
    // prefetch next step's x (latency hidden under WMMAs)
    {
      int stepn = (step + 1 < TT) ? (step + 1) : step;
      int t2 = dir ? (TT - 1 - stepn) : stepn;
      const float* xp = xbase + (size_t)t2 * (AA * FF);
      #pragma unroll
      for (int k = 0; k < FF; ++k) xn[k] = xp[k];
    }

    // A fragments for h (from LDS, 16B vector loads; k-runs contiguous)
    V16 ah0, ah1;
    {
      const _Float16* hp = &hbuf[nl * HH];
      int o0 = 8 * hi;
      ah0.f4[0] = *(const float4*)(hp + o0);
      ah0.f4[1] = *(const float4*)(hp + o0 + 16);
      ah1.f4[0] = *(const float4*)(hp + 32 + o0);
      ah1.f4[1] = *(const float4*)(hp + 32 + o0 + 16);
    }

    V8F ar, az, anh, ani;
    #pragma unroll
    for (int v = 0; v < 8; ++v) { ar.f[v] = bias_r; az.f[v] = bias_z; anh.f[v] = bias_nh; ani.f[v] = bias_ni; }

    // r gate: h-path (2 ksteps) + x-path
    ar.v = __builtin_amdgcn_wmma_f32_16x16x32_f16(false, ah0.v, false, bh[0][0].v, (short)0, ar.v, false, false);
    ar.v = __builtin_amdgcn_wmma_f32_16x16x32_f16(false, ah1.v, false, bh[0][1].v, (short)0, ar.v, false, false);
    ar.v = __builtin_amdgcn_wmma_f32_16x16x32_f16(false, axf.v, false, bx[0].v,    (short)0, ar.v, false, false);
    // z gate
    az.v = __builtin_amdgcn_wmma_f32_16x16x32_f16(false, ah0.v, false, bh[1][0].v, (short)0, az.v, false, false);
    az.v = __builtin_amdgcn_wmma_f32_16x16x32_f16(false, ah1.v, false, bh[1][1].v, (short)0, az.v, false, false);
    az.v = __builtin_amdgcn_wmma_f32_16x16x32_f16(false, axf.v, false, bx[1].v,    (short)0, az.v, false, false);
    // n gate: keep x-path (i_n) and h-path (h_n) separate (r multiplies only h_n)
    anh.v = __builtin_amdgcn_wmma_f32_16x16x32_f16(false, ah0.v, false, bh[2][0].v, (short)0, anh.v, false, false);
    anh.v = __builtin_amdgcn_wmma_f32_16x16x32_f16(false, ah1.v, false, bh[2][1].v, (short)0, anh.v, false, false);
    ani.v = __builtin_amdgcn_wmma_f32_16x16x32_f16(false, axf.v, false, bx[2].v,    (short)0, ani.v, false, false);

    // gate nonlinearities (C/D layout: row m = v + 8*hi, col = gcol); h_prev from registers
    #pragma unroll
    for (int v = 0; v < 8; ++v) {
      float r = fast_sigmoid(ar.f[v]);
      float z = fast_sigmoid(az.f[v]);
      float n = fast_tanh(ani.f[v] + r * anh.f[v]);
      hreg[v] = (1.f - z) * n + z * hreg[v];
    }
    __syncthreads();              // all waves done reading hbuf
    #pragma unroll
    for (int v = 0; v < 8; ++v) {
      int m = v + 8 * hi;
      _Float16 hv = (_Float16)hreg[v];
      hbuf[m * HH + gcol] = hv;
      int ba = rowbase + m;
      enc16[((size_t)ba * TT + t) * EH + dir * HH + gcol] = hv;
    }
    #pragma unroll
    for (int k = 0; k < FF; ++k) xv[k] = xn[k];
    __syncthreads();
  }
}

// ---------------- attention projection: E = enc_out @ W_e^T ----------------
// M = BA*TT rows, K = 128, N = 12 (padded 16). Block 256 = 8 waves, 16 rows/wave.
__global__ __launch_bounds__(256) void attnE_kernel(const _Float16* __restrict__ enc16,
                                                    const _Float16* __restrict__ Wattn16,
                                                    float* __restrict__ E) {
  const int wv = threadIdx.x >> 5;
  const int ln = threadIdx.x & 31;
  const int hi = ln >> 4;
  const int nl = ln & 15;
  const size_t rowbase = (size_t)blockIdx.x * 128 + wv * 16;

  V16 bf[4];
  #pragma unroll
  for (int ks = 0; ks < 4; ++ks)
    #pragma unroll
    for (int e = 0; e < 16; ++e)
      bf[ks].h[e] = Wattn16[nl * EH + 32 * ks + kmapB(e, hi)];

  V8F acc;
  #pragma unroll
  for (int v = 0; v < 8; ++v) acc.f[v] = 0.f;

  #pragma unroll
  for (int ks = 0; ks < 4; ++ks) {
    V16 a;
    const _Float16* p = enc16 + (rowbase + nl) * EH + 32 * ks + 8 * hi;
    a.f4[0] = *(const float4*)(p);
    a.f4[1] = *(const float4*)(p + 16);
    acc.v = __builtin_amdgcn_wmma_f32_16x16x32_f16(false, a.v, false, bf[ks].v, (short)0, acc.v, false, false);
  }
  #pragma unroll
  for (int v = 0; v < 8; ++v) {
    int m = v + 8 * hi;
    if (nl < DD) E[(rowbase + m) * DD + nl] = acc.f[v];
  }
}

// ---------------- decoder init: h_dec0 and d_in0 ----------------
// grid 1024 x 128 threads; each wave handles one ba.
__global__ __launch_bounds__(128) void dec_init_kernel(const float* __restrict__ src,
                                                       const _Float16* __restrict__ enc16,
                                                       const float* __restrict__ W_e2d,
                                                       const float* __restrict__ b_e2d,
                                                       float* __restrict__ h_dec,
                                                       float* __restrict__ d_in) {
  const int ba = blockIdx.x * 4 + (threadIdx.x >> 5);
  const int ln = threadIdx.x & 31;
  if (ln < DD) {
    float s = b_e2d[ln];
    const float* w = W_e2d + ln * EH;
    const _Float16* ef = enc16 + ((size_t)ba * TT + (TT - 1)) * EH;  // hf_last: t=T-1, cols 0..63
    const _Float16* eb = enc16 + ((size_t)ba * TT + 0) * EH;         // hb_last: t=0,   cols 64..127
    #pragma unroll 8
    for (int e = 0; e < HH; ++e) s += (float)ef[e] * w[e];
    #pragma unroll 8
    for (int e = 0; e < HH; ++e) s += (float)eb[HH + e] * w[HH + e];
    h_dec[ba * DD + ln] = s;
  }
  if (ln < FF) {
    int b = ba >> 5, a = ba & 31;
    d_in[ba * FF + ln] = src[((size_t)b * TT + (TT - 1)) * (AA * FF) + a * FF + ln];
  }
}

// ---------------- one decoder step (launched P times) ----------------
// grid = BA, block = 128
__global__ __launch_bounds__(128) void dec_step_kernel(
    const _Float16* __restrict__ enc16, const float* __restrict__ E,
    const float* __restrict__ W_attn, const float* __restrict__ b_attn,
    const float* __restrict__ v_attn,
    const float* __restrict__ Wih_d, const float* __restrict__ Whh_d,
    const float* __restrict__ bih_d, const float* __restrict__ bhh_d,
    const float* __restrict__ W_out, const float* __restrict__ b_out,
    float* __restrict__ h_dec, float* __restrict__ d_in,
    float* __restrict__ out, int p, int P) {
  const int b = blockIdx.x;
  const int tid = threadIdx.x;
  __shared__ float q[DD];
  __shared__ float sc[TT];
  __shared__ float ctxs[EH];
  __shared__ float gi36[3 * DD];
  __shared__ float gh36[3 * DD];
  __shared__ float hnew[DD];
  __shared__ float red[2];

  if (tid < DD) {                       // q = h @ W_h^T + b_attn  (hid part of W_attn)
    float s = b_attn[tid];
    const float* w = W_attn + tid * (2 * HH + DD);
    #pragma unroll
    for (int k = 0; k < DD; ++k) s += h_dec[b * DD + k] * w[k];
    q[tid] = s;
  }
  __syncthreads();
  if (tid < TT) {                       // scores
    float s = 0.f;
    const float* Ep = E + ((size_t)b * TT + tid) * DD;
    #pragma unroll
    for (int d = 0; d < DD; ++d) s += v_attn[d] * fast_tanh(Ep[d] + q[d]);
    sc[tid] = s;
  }
  __syncthreads();
  if (tid == 0) { float m = sc[0]; for (int i = 1; i < TT; ++i) m = fmaxf(m, sc[i]); red[0] = m; }
  __syncthreads();
  if (tid < TT) sc[tid] = __builtin_amdgcn_exp2f(1.4426950408889634f * (sc[tid] - red[0]));
  __syncthreads();
  if (tid == 0) { float s = 0.f; for (int i = 0; i < TT; ++i) s += sc[i]; red[1] = __builtin_amdgcn_rcpf(s); }
  __syncthreads();
  {                                     // ctx = softmax(scores) . enc_out
    float inv = red[1];
    float c = 0.f;
    const _Float16* ep = enc16 + (size_t)b * TT * EH + tid;
    #pragma unroll 8
    for (int t = 0; t < TT; ++t) c += sc[t] * (float)ep[t * EH];
    ctxs[tid] = c * inv;
  }
  __syncthreads();
  if (tid < 3 * DD) {                   // decoder GRU gates, rnn_in = [d_in, ctx]
    float gi = bih_d[tid];
    const float* wi = Wih_d + tid * (FF + EH);
    #pragma unroll
    for (int k = 0; k < FF; ++k) gi += d_in[b * FF + k] * wi[k];
    #pragma unroll 8
    for (int k = 0; k < EH; ++k) gi += ctxs[k] * wi[FF + k];
    float gh = bhh_d[tid];
    const float* wh = Whh_d + tid * DD;
    #pragma unroll
    for (int k = 0; k < DD; ++k) gh += h_dec[b * DD + k] * wh[k];
    gi36[tid] = gi; gh36[tid] = gh;
  }
  __syncthreads();
  if (tid < DD) {
    float r = fast_sigmoid(gi36[tid] + gh36[tid]);
    float z = fast_sigmoid(gi36[DD + tid] + gh36[DD + tid]);
    float n = fast_tanh(gi36[2 * DD + tid] + r * gh36[2 * DD + tid]);
    hnew[tid] = (1.f - z) * n + z * h_dec[b * DD + tid];
  }
  __syncthreads();
  if (tid < DD) h_dec[b * DD + tid] = hnew[tid];
  if (tid < FF) {                       // pred = h2 @ W_out^T + b_out
    float s = b_out[tid];
    const float* w = W_out + tid * DD;
    #pragma unroll
    for (int d = 0; d < DD; ++d) s += hnew[d] * w[d];
    d_in[b * FF + tid] = s;
    int bb = b >> 5, a = b & 31;        // out[bb][p][a*F+f]
    out[((size_t)bb * P + p) * (AA * FF) + a * FF + tid] = s;
  }
}

extern "C" void kernel_launch(void* const* d_in, const int* in_sizes, int n_in,
                              void* d_out, int out_size, void* d_ws, size_t ws_size,
                              hipStream_t stream) {
  const float* src    = (const float*)d_in[0];
  // d_in[1] = pred_len (device scalar); derive P from out_size instead (no sync allowed)
  const float* Wih_f  = (const float*)d_in[2];
  const float* Whh_f  = (const float*)d_in[3];
  const float* bih_f  = (const float*)d_in[4];
  const float* bhh_f  = (const float*)d_in[5];
  const float* Wih_b  = (const float*)d_in[6];
  const float* Whh_b  = (const float*)d_in[7];
  const float* bih_b  = (const float*)d_in[8];
  const float* bhh_b  = (const float*)d_in[9];
  const float* W_attn = (const float*)d_in[10];
  const float* b_attn = (const float*)d_in[11];
  const float* v_attn = (const float*)d_in[12];
  const float* W_e2d  = (const float*)d_in[13];
  const float* b_e2d  = (const float*)d_in[14];
  const float* Wih_d  = (const float*)d_in[15];
  const float* Whh_d  = (const float*)d_in[16];
  const float* bih_d  = (const float*)d_in[17];
  const float* bhh_d  = (const float*)d_in[18];
  const float* W_out  = (const float*)d_in[19];
  const float* b_out  = (const float*)d_in[20];
  float* out = (float*)d_out;

  int P = out_size / (BB * AA * FF);
  if (P <= 0) P = 12;

  // ---- workspace carving (all chunks 256B-aligned) ----
  char* w = (char*)d_ws;
  _Float16* enc16 = (_Float16*)w; w += (size_t)BA * TT * EH * 2;   // 67,108,864 B
  float* E        = (float*)w;    w += (size_t)BA * TT * DD * 4;   // 12,582,912 B
  float* h_dec    = (float*)w;    w += (size_t)BA * DD * 4;        //    196,608 B
  float* d_inb    = (float*)w;    w += (size_t)BA * FF * 4;        //     98,304 B
  _Float16* Whh16f = (_Float16*)w; w += G3H * HH * 2;              //     24,576 B
  _Float16* Whh16b = (_Float16*)w; w += G3H * HH * 2;
  _Float16* Wih16f = (_Float16*)w; w += G3H * 8 * 2;               //      3,072 B
  _Float16* Wih16b = (_Float16*)w; w += G3H * 8 * 2;
  _Float16* Wattn16 = (_Float16*)w; w += 16 * EH * 2;              //      4,096 B
  (void)ws_size; (void)in_sizes; (void)n_in;

  // 1) weight precision prep
  prep_kernel<<<48, 256, 0, stream>>>(Whh_f, Whh_b, Wih_f, Wih_b, W_attn,
                                      Whh16f, Whh16b, Wih16f, Wih16b, Wattn16);
  // 2) bidirectional GRU encoder (WMMA recurrent core)
  enc_kernel<<<dim3(BA / 16, 2), 128, 0, stream>>>(src, Whh16f, Whh16b, Wih16f, Wih16b,
                                                   bih_f, bhh_f, bih_b, bhh_b, enc16);
  // 3) decoder-step-invariant attention projection (WMMA GEMM)
  attnE_kernel<<<(BA * TT) / 128, 256, 0, stream>>>(enc16, Wattn16, E);
  // 4) decoder initial state
  dec_init_kernel<<<BA / 4, 128, 0, stream>>>(src, enc16, W_e2d, b_e2d, h_dec, d_inb);
  // 5) P decoder steps
  for (int p = 0; p < P; ++p) {
    dec_step_kernel<<<BA, 128, 0, stream>>>(enc16, E, W_attn, b_attn, v_attn,
                                            Wih_d, Whh_d, bih_d, bhh_d, W_out, b_out,
                                            h_dec, d_inb, out, p, P);
  }
}
